// AttHead_51599737094903
// MI455X (gfx1250) — matrix operand order repaired and checked by reading, not compiled, not run on hardware
//
#include <hip/hip_runtime.h>

#define B_ 512
#define T_ 256
#define C_ 384
#define H_ 64
#define SCALE_ 0.051031036307982884f  // 384^-0.5

typedef __attribute__((ext_vector_type(16))) __bf16 v16bf;
typedef __attribute__((ext_vector_type(8)))  float  v8f;

struct __align__(16) U128 { unsigned int x, y, z, w; };

union Frag {
  v16bf v;
  unsigned short us[16];
  U128 q[2];
};

// Native f32 -> bf16 (RNE) via hardware convert (lowers to v_cvt_pk_bf16_f32).
__device__ __forceinline__ unsigned short f2bf(float f) {
  return __builtin_bit_cast(unsigned short, (__bf16)f);
}

__device__ __forceinline__ v8f wmma_bf16(v16bf a, v16bf b, v8f c) {
  return __builtin_amdgcn_wmma_f32_16x16x32_bf16(false, a, false, b, (short)0, c,
                                                 false, false);
}

__device__ __forceinline__ float grpmax16(float v) {
  v = fmaxf(v, __shfl_xor(v, 1, 32));
  v = fmaxf(v, __shfl_xor(v, 2, 32));
  v = fmaxf(v, __shfl_xor(v, 4, 32));
  v = fmaxf(v, __shfl_xor(v, 8, 32));
  return v;
}
__device__ __forceinline__ float grpsum16(float v) {
  v += __shfl_xor(v, 1, 32);
  v += __shfl_xor(v, 2, 32);
  v += __shfl_xor(v, 4, 32);
  v += __shfl_xor(v, 8, 32);
  return v;
}

__global__ __launch_bounds__(256) void attn_head_fused(
    const float* __restrict__ x, const float* __restrict__ Wk,
    const float* __restrict__ Wq, const float* __restrict__ Wv,
    float* __restrict__ out) {
  // Padded so 16-lane fragment loads hit 16 distinct bank groups.
  __shared__ __align__(16) unsigned short Wt[3][H_][392];  // W^T bf16: [q,k,v]
  __shared__ __align__(16) unsigned short KsS[T_][72];     // K  [token][head]
  __shared__ __align__(16) unsigned short VtS[H_][264];    // V^T [head][token]
  __shared__ __align__(16) unsigned short Sw[8][16][72];   // per-wave transpose scratch

  const int tid = threadIdx.x;
  const int lane = tid & 31;
  const int w = tid >> 5;
  const int lm = lane & 15;
  const int lhalf = lane >> 4;
  const int koff = lhalf * 8;
  const int b = blockIdx.x;
  const float* xb = x + (size_t)b * T_ * C_;

  const int rts[2] = { w, 15 - w };  // causal-balanced row-tile pair
  Frag qfrag[2][2];                  // [rowtile][kstep] A-fragments of Q

  // ---- stage all three W^T matrices (bf16) into LDS -----------------------
  {
    const float* Wsrc[3] = { Wq, Wk, Wv };
#pragma unroll
    for (int m = 0; m < 3; m++) {
      const float* Wg = Wsrc[m];
      for (int idx = tid; idx < (H_ * C_) / 4; idx += 256) {
        const int h4 = (idx & 15) * 4;  // lane-varying h -> coalesced float4 reads
        const int c = idx >> 4;
        const float4 f = *(const float4*)(Wg + (size_t)c * H_ + h4);
        Wt[m][h4 + 0][c] = f2bf(f.x);
        Wt[m][h4 + 1][c] = f2bf(f.y);
        Wt[m][h4 + 2][c] = f2bf(f.z);
        Wt[m][h4 + 3][c] = f2bf(f.w);
      }
    }
  }
  __syncthreads();

  // ---- single pass over x: Q, K, V projections together -------------------
#pragma unroll 1
  for (int t = 0; t < 2; t++) {
    const int rt = rts[t];
    const float* xrow = xb + (size_t)(rt * 16 + lm) * C_;
    v8f acc[3][4];
#pragma unroll
    for (int m = 0; m < 3; m++)
#pragma unroll
      for (int cb = 0; cb < 4; cb++) acc[m][cb] = (v8f)0.0f;

    for (int kk = 0; kk < C_; kk += 32) {
      // pack one x A-fragment (f32 -> bf16), reused by all 12 WMMAs
      Frag a;
      const float4 f0 = *(const float4*)(xrow + kk + lhalf * 16);
      const float4 f1 = *(const float4*)(xrow + kk + lhalf * 16 + 4);
      const float4 f2 = *(const float4*)(xrow + kk + lhalf * 16 + 8);
      const float4 f3 = *(const float4*)(xrow + kk + lhalf * 16 + 12);
      a.v[0] = (__bf16)f0.x; a.v[1] = (__bf16)f0.y; a.v[2] = (__bf16)f0.z; a.v[3] = (__bf16)f0.w;
      a.v[4] = (__bf16)f1.x; a.v[5] = (__bf16)f1.y; a.v[6] = (__bf16)f1.z; a.v[7] = (__bf16)f1.w;
      a.v[8] = (__bf16)f2.x; a.v[9] = (__bf16)f2.y; a.v[10] = (__bf16)f2.z; a.v[11] = (__bf16)f2.w;
      a.v[12] = (__bf16)f3.x; a.v[13] = (__bf16)f3.y; a.v[14] = (__bf16)f3.z; a.v[15] = (__bf16)f3.w;
#pragma unroll
      for (int m = 0; m < 3; m++) {
        // batch the 4 B-fragment loads into independent regs, then 4 WMMAs
        Frag bfr[4];
#pragma unroll
        for (int cb = 0; cb < 4; cb++) {
          const U128* bp = (const U128*)&Wt[m][cb * 16 + lm][kk + lhalf * 16];
          bfr[cb].q[0] = bp[0]; bfr[cb].q[1] = bp[1];
        }
#pragma unroll
        for (int cb = 0; cb < 4; cb++)
          acc[m][cb] = wmma_bf16(a.v, bfr[cb].v, acc[m][cb]);
      }
    }

    // Q: C-layout -> A-layout via wave-private LDS scratch, keep in registers
#pragma unroll
    for (int cb = 0; cb < 4; cb++)
#pragma unroll
      for (int r = 0; r < 8; r++)
        Sw[w][r + lhalf * 8][cb * 16 + lm] = f2bf(acc[0][cb][r]);
#pragma unroll
    for (int q = 0; q < 2; q++) {
      qfrag[t][q].q[0] = *(const U128*)&Sw[w][lm][q * 32 + lhalf * 16];
      qfrag[t][q].q[1] = *(const U128*)&Sw[w][lm][q * 32 + lhalf * 16 + 8];
    }
    // K -> KsS [token][head]
#pragma unroll
    for (int cb = 0; cb < 4; cb++)
#pragma unroll
      for (int r = 0; r < 8; r++)
        KsS[rt * 16 + r + lhalf * 8][cb * 16 + lm] = f2bf(acc[1][cb][r]);
    // V -> VtS [head][token] (transposed for contiguous B-fragments)
#pragma unroll
    for (int cb = 0; cb < 4; cb++)
#pragma unroll
      for (int r = 0; r < 8; r++)
        VtS[cb * 16 + lm][rt * 16 + r + lhalf * 8] = f2bf(acc[2][cb][r]);
  }
  __syncthreads();

  // ---------------- fused causal flash attention ---------------------------
#pragma unroll 1
  for (int t = 0; t < 2; t++) {
    const int i = rts[t];
    const int rowbase = i * 16;
    v8f oacc[4];
#pragma unroll
    for (int cb = 0; cb < 4; cb++) oacc[cb] = (v8f)0.0f;
    float rmax[8], rsum[8];
#pragma unroll
    for (int r = 0; r < 8; r++) { rmax[r] = -__builtin_inff(); rsum[r] = 0.0f; }

    const int npairs = (i + 2) >> 1;  // key tiles 0..i, two at a time
    for (int jj = 0; jj < npairs; jj++) {
      const int jbase = jj * 32;
      const bool haveB = (2 * jj + 1) <= i;  // wave-uniform
      // batch all K-fragment loads, then the S WMMAs
      Frag kf[4];
#pragma unroll
      for (int q = 0; q < 2; q++) {
        const U128* kp = (const U128*)&KsS[jbase + lm][q * 32 + lhalf * 16];
        kf[q].q[0] = kp[0]; kf[q].q[1] = kp[1];
      }
      if (haveB) {
#pragma unroll
        for (int q = 0; q < 2; q++) {
          const U128* kp = (const U128*)&KsS[jbase + 16 + lm][q * 32 + lhalf * 16];
          kf[2 + q].q[0] = kp[0]; kf[2 + q].q[1] = kp[1];
        }
      }
      v8f sa = (v8f)0.0f, sb = (v8f)0.0f;
      sa = wmma_bf16(qfrag[t][0].v, kf[0].v, sa);
      sa = wmma_bf16(qfrag[t][1].v, kf[1].v, sa);
      if (haveB) {
        sb = wmma_bf16(qfrag[t][0].v, kf[2].v, sb);
        sb = wmma_bf16(qfrag[t][1].v, kf[3].v, sb);
      }
      // issue V-fragment loads now; softmax VALU below hides their latency
      Frag vf[4];
#pragma unroll
      for (int cb = 0; cb < 4; cb++) {
        const U128* vp = (const U128*)&VtS[cb * 16 + lm][jbase + lhalf * 16];
        vf[cb].q[0] = vp[0]; vf[cb].q[1] = vp[1];
      }
      const int col0 = jbase + lm;
      const int col1 = jbase + 16 + lm;
#pragma unroll
      for (int r = 0; r < 8; r++) {
        const int rowg = rowbase + r + lhalf * 8;  // C-layout M index
        float v0 = sa[r] * SCALE_;
        if (col0 > rowg) v0 = -__builtin_inff();
        float v1 = haveB ? sb[r] * SCALE_ : -__builtin_inff();
        if (col1 > rowg) v1 = -__builtin_inff();
        const float mx = grpmax16(fmaxf(v0, v1));
        const float mnew = fmaxf(rmax[r], mx);
        const float corr = __expf(rmax[r] - mnew);
        const float p0 = __expf(v0 - mnew);
        const float p1 = __expf(v1 - mnew);
        rsum[r] = rsum[r] * corr + grpsum16(p0 + p1);
        rmax[r] = mnew;
#pragma unroll
        for (int cb = 0; cb < 4; cb++) oacc[cb][r] *= corr;
        Sw[w][r + lhalf * 8][lm] = f2bf(p0);       // P -> scratch for transpose
        Sw[w][r + lhalf * 8][16 + lm] = f2bf(p1);
      }
      Frag pf;  // P as A-fragment (16 rows x 32 keys)
      pf.q[0] = *(const U128*)&Sw[w][lm][lhalf * 16];
      pf.q[1] = *(const U128*)&Sw[w][lm][lhalf * 16 + 8];
#pragma unroll
      for (int cb = 0; cb < 4; cb++)  // O += P @ V
        oacc[cb] = wmma_bf16(pf.v, vf[cb].v, oacc[cb]);
    }
    // epilogue: normalize and store f32
#pragma unroll
    for (int r = 0; r < 8; r++) {
      const float inv = 1.0f / rsum[r];
      const int rowg = rowbase + r + lhalf * 8;
      float* orow = out + ((size_t)b * T_ + rowg) * H_;
#pragma unroll
      for (int cb = 0; cb < 4; cb++)
        orow[cb * 16 + lm] = oacc[cb][r] * inv;
    }
  }
}

extern "C" void kernel_launch(void* const* d_in, const int* in_sizes, int n_in,
                              void* d_out, int out_size, void* d_ws, size_t ws_size,
                              hipStream_t stream) {
  (void)in_sizes; (void)n_in; (void)out_size; (void)d_ws; (void)ws_size;
  const float* x  = (const float*)d_in[0];
  const float* Wk = (const float*)d_in[1];
  const float* Wq = (const float*)d_in[2];
  const float* Wv = (const float*)d_in[3];
  float* out = (float*)d_out;
  attn_head_fused<<<dim3(B_), dim3(256), 0, stream>>>(x, Wk, Wq, Wv, out);
}
